// HybridAttentionRAG_71253507441107
// MI455X (gfx1250) — matrix operand order repaired and checked by reading, not compiled, not run on hardware
//
#include <hip/hip_runtime.h>
#include <math.h>

// ---------------- problem constants ----------------
#define S_LEN   4480
#define H_DIM   1024
#define NHEAD   16
#define HD      64
#define GHEAD   4
#define HG      256
#define NLM     32
#define NSEG    4
#define LSEG    256
#define WWIN    512
#define WSTRIDE 448
#define NWIN    10
#define FFDIM   4096

// ---------------- WMMA types / helpers ----------------
typedef __attribute__((ext_vector_type(16))) __bf16 v16bf;
typedef __attribute__((ext_vector_type(8)))  float  v8f;

__device__ __forceinline__ unsigned short f2bf(float f) {
  unsigned int u = __float_as_uint(f);
  u += 0x7FFFu + ((u >> 16) & 1u);          // round-to-nearest-even
  return (unsigned short)(u >> 16);
}
__device__ __forceinline__ unsigned int pack2(float a, float b) {
  return (unsigned int)f2bf(a) | ((unsigned int)f2bf(b) << 16);
}
union FragBF { uint4 q[2]; v16bf v; };

__device__ __forceinline__ v8f v8f_zero() {
  v8f z;
#pragma unroll
  for (int i = 0; i < 8; ++i) z[i] = 0.f;
  return z;
}
__device__ __forceinline__ v8f wmma_bf16(const FragBF& a, const FragBF& b, v8f c) {
  return __builtin_amdgcn_wmma_f32_16x16x32_bf16(false, a.v, false, b.v,
                                                 (short)0, c, false, false);
}

// ---------------- GEMM: C[M,N] = A[M,K] * W[N,K]^T + bias (+res, relu) ----------------
#define BM 128
#define BN 128
#define BK 32
#define LDT 40      // padded ushort row stride (80B -> conflict-free, 16B aligned)

__global__ __launch_bounds__(256)
void gemm_wmma(const float* __restrict__ A, int lda,
               const float* __restrict__ W, int ldw,
               const float* __restrict__ bias,
               const float* __restrict__ res,
               float* __restrict__ C, int ldc,
               int M, int N, int K, int doRelu)
{
  __shared__ unsigned short As[BM * LDT];
  __shared__ unsigned short Bs[BN * LDT];

  const int bm = blockIdx.x * BM;
  const int bn = blockIdx.y * BN;
  const int t    = threadIdx.x;
  const int wave = t >> 5;
  const int lane = t & 31;
  const int wm = (wave >> 1) * 32;   // 0,32,64,96
  const int wn = (wave & 1) * 64;    // 0,64
  const int hi = lane >> 4, l15 = lane & 15;

  v8f acc[2][4];
#pragma unroll
  for (int i = 0; i < 2; ++i)
#pragma unroll
    for (int j = 0; j < 4; ++j) acc[i][j] = v8f_zero();

  const int lrow = t >> 1;           // 0..127
  const int lcol = (t & 1) * 16;     // 0 / 16
  int aRow = bm + lrow; if (aRow > M - 1) aRow = M - 1;
  const bool aLive = (bm + lrow) < M;
  const float* aPtr = A + (size_t)aRow * lda + lcol;
  const float* bPtr = W + (size_t)(bn + lrow) * ldw + lcol;
  unsigned short* aDst = &As[lrow * LDT + lcol];
  unsigned short* bDst = &Bs[lrow * LDT + lcol];

  for (int k0 = 0; k0 < K; k0 += BK) {
    if (k0 + BK < K) {                       // prefetch next K tiles into caches
      __builtin_prefetch(aPtr + BK, 0, 1);
      __builtin_prefetch(bPtr + BK, 0, 1);
    }
    {
      const float4* s = (const float4*)aPtr;
      float4 f0 = s[0], f1 = s[1], f2 = s[2], f3 = s[3];
      if (!aLive) { f0 = f1 = f2 = f3 = make_float4(0.f, 0.f, 0.f, 0.f); }
      uint4 o0, o1;
      o0.x = pack2(f0.x, f0.y); o0.y = pack2(f0.z, f0.w);
      o0.z = pack2(f1.x, f1.y); o0.w = pack2(f1.z, f1.w);
      o1.x = pack2(f2.x, f2.y); o1.y = pack2(f2.z, f2.w);
      o1.z = pack2(f3.x, f3.y); o1.w = pack2(f3.z, f3.w);
      *(uint4*)aDst = o0; *(uint4*)(aDst + 8) = o1;
    }
    {
      const float4* s = (const float4*)bPtr;
      float4 f0 = s[0], f1 = s[1], f2 = s[2], f3 = s[3];
      uint4 o0, o1;
      o0.x = pack2(f0.x, f0.y); o0.y = pack2(f0.z, f0.w);
      o0.z = pack2(f1.x, f1.y); o0.w = pack2(f1.z, f1.w);
      o1.x = pack2(f2.x, f2.y); o1.y = pack2(f2.z, f2.w);
      o1.z = pack2(f3.x, f3.y); o1.w = pack2(f3.z, f3.w);
      *(uint4*)bDst = o0; *(uint4*)(bDst + 8) = o1;
    }
    __syncthreads();

    FragBF af[2], bf_[4];
#pragma unroll
    for (int i = 0; i < 2; ++i) {
      const unsigned short* p = &As[(wm + 16 * i + l15) * LDT];
      af[i].q[0] = *(const uint4*)(p + hi * 8);
      af[i].q[1] = *(const uint4*)(p + 16 + hi * 8);
    }
#pragma unroll
    for (int j = 0; j < 4; ++j) {
      const unsigned short* p = &Bs[(wn + 16 * j + l15) * LDT];
      bf_[j].q[0] = *(const uint4*)(p + hi * 8);
      bf_[j].q[1] = *(const uint4*)(p + 16 + hi * 8);
    }
#pragma unroll
    for (int i = 0; i < 2; ++i)
#pragma unroll
      for (int j = 0; j < 4; ++j)
        acc[i][j] = wmma_bf16(af[i], bf_[j], acc[i][j]);

    __syncthreads();
    aPtr += BK; bPtr += BK;
  }

#pragma unroll
  for (int i = 0; i < 2; ++i) {
#pragma unroll
    for (int j = 0; j < 4; ++j) {
      const int col = bn + wn + 16 * j + l15;
      const float bval = bias[col];
#pragma unroll
      for (int r = 0; r < 8; ++r) {
        const int row = bm + wm + 16 * i + 8 * hi + r;
        if (row < M) {
          float v = acc[i][j][r] + bval;
          if (res) v += res[(size_t)row * ldc + col];
          if (doRelu) v = fmaxf(v, 0.f);
          C[(size_t)row * ldc + col] = v;
        }
      }
    }
  }
}

// ---------------- flash attention (head_dim=64, WMMA both GEMMs) ----------------
#define FQT 64
#define FKC 64
#define FLD 72    // ushort stride (144B, conflict-free, 16B aligned)
#define SLD 65    // float stride for O accumulator

__global__ __launch_bounds__(128)
void flash_attn(const float* __restrict__ Q, int ldq, long long qB,
                const float* __restrict__ Km, int ldk, long long kB,
                const float* __restrict__ Vm, int ldv, long long vB,
                float* __restrict__ O, int ldo, long long oB,
                int kvLen, int validLen, float scale)
{
  __shared__ unsigned short Qs[FQT * FLD];
  __shared__ unsigned short Ks[FKC * FLD];
  __shared__ unsigned short Vt[HD  * FLD];   // V transposed: Vt[d][kk]
  __shared__ unsigned short Ps[FQT * FLD];   // softmaxed probs (bf16)
  __shared__ float Ol[FQT * SLD];
  __shared__ float lArr[FQT];

  const int q0    = blockIdx.x * FQT;
  const int head  = blockIdx.y;
  const int batch = blockIdx.z;
  const int t     = threadIdx.x;
  const int wave  = t >> 5;
  const int lane  = t & 31;
  const int hi = lane >> 4, l15 = lane & 15;
  const int wb = wave * 16;                  // wave's q-row strip

  const float* Qb = Q  + (long long)batch * qB + head * HD;
  const float* Kb = Km + (long long)batch * kB + head * HD;
  const float* Vb = Vm + (long long)batch * vB + head * HD;
  float*       Ob = O  + (long long)batch * oB + head * HD;

  {   // load Q tile (bf16) and zero O accumulator
    const int row = t >> 1, c0 = (t & 1) * 32;
    const float4* s = (const float4*)(Qb + (size_t)(q0 + row) * ldq + c0);
#pragma unroll
    for (int v4 = 0; v4 < 8; ++v4) {
      float4 f = s[v4];
      *(uint2*)&Qs[row * FLD + c0 + v4 * 4] =
          make_uint2(pack2(f.x, f.y), pack2(f.z, f.w));
    }
    for (int c = c0; c < c0 + 32; ++c) Ol[row * SLD + c] = 0.f;
  }
  float m_r[8], l_r[8];
#pragma unroll
  for (int r = 0; r < 8; ++r) { m_r[r] = -1e30f; l_r[r] = 0.f; }
  __syncthreads();

  for (int c0 = 0; c0 < kvLen; c0 += FKC) {
    {   // load K chunk; load V chunk transposed
      const int row = t >> 1, cc = (t & 1) * 32;
      const float4* sk = (const float4*)(Kb + (size_t)(c0 + row) * ldk + cc);
#pragma unroll
      for (int v4 = 0; v4 < 8; ++v4) {
        float4 f = sk[v4];
        *(uint2*)&Ks[row * FLD + cc + v4 * 4] =
            make_uint2(pack2(f.x, f.y), pack2(f.z, f.w));
      }
      const float4* sv = (const float4*)(Vb + (size_t)(c0 + row) * ldv + cc);
#pragma unroll
      for (int v4 = 0; v4 < 8; ++v4) {
        float4 f = sv[v4];
        const int d = cc + v4 * 4;
        Vt[(d + 0) * FLD + row] = f2bf(f.x);
        Vt[(d + 1) * FLD + row] = f2bf(f.y);
        Vt[(d + 2) * FLD + row] = f2bf(f.z);
        Vt[(d + 3) * FLD + row] = f2bf(f.w);
      }
    }
    __syncthreads();

    // S = Q*K^T for this wave's 16 rows (4 n-tiles x 2 k-steps)
    v8f sacc[4];
#pragma unroll
    for (int j = 0; j < 4; ++j) sacc[j] = v8f_zero();
#pragma unroll
    for (int kd = 0; kd < HD; kd += 32) {
      FragBF a;
      const unsigned short* pa = &Qs[(wb + l15) * FLD + kd];
      a.q[0] = *(const uint4*)(pa + hi * 8);
      a.q[1] = *(const uint4*)(pa + 16 + hi * 8);
#pragma unroll
      for (int j = 0; j < 4; ++j) {
        FragBF b;
        const unsigned short* pb = &Ks[(16 * j + l15) * FLD + kd];
        b.q[0] = *(const uint4*)(pb + hi * 8);
        b.q[1] = *(const uint4*)(pb + 16 + hi * 8);
        sacc[j] = wmma_bf16(a, b, sacc[j]);
      }
    }
    // scale + key-validity mask in registers
#pragma unroll
    for (int j = 0; j < 4; ++j) {
      const bool ok = (c0 + 16 * j + l15) < validLen;
#pragma unroll
      for (int r = 0; r < 8; ++r)
        sacc[j][r] = ok ? sacc[j][r] * scale : -1e30f;
    }
    // online softmax (row = wb + 8*hi + r, reduce across the 16 column lanes)
    float corr[8];
#pragma unroll
    for (int r = 0; r < 8; ++r) {
      float cm = -1e30f;
#pragma unroll
      for (int j = 0; j < 4; ++j) cm = fmaxf(cm, sacc[j][r]);
      for (int d = 1; d < 16; d <<= 1) cm = fmaxf(cm, __shfl_xor(cm, d, 32));
      const float nm = fmaxf(m_r[r], cm);
      corr[r] = __expf(m_r[r] - nm);
      float ls = 0.f;
      const int row = wb + 8 * hi + r;
#pragma unroll
      for (int j = 0; j < 4; ++j) {
        const float p = __expf(sacc[j][r] - nm);
        ls += p;
        Ps[row * FLD + 16 * j + l15] = f2bf(p);
      }
      for (int d = 1; d < 16; d <<= 1) ls += __shfl_xor(ls, d, 32);
      l_r[r] = l_r[r] * corr[r] + ls;
      m_r[r] = nm;
    }
    __syncthreads();

    // D = P*V
    v8f dacc[4];
#pragma unroll
    for (int j = 0; j < 4; ++j) dacc[j] = v8f_zero();
#pragma unroll
    for (int kk = 0; kk < FKC; kk += 32) {
      FragBF a;
      const unsigned short* pa = &Ps[(wb + l15) * FLD + kk];
      a.q[0] = *(const uint4*)(pa + hi * 8);
      a.q[1] = *(const uint4*)(pa + 16 + hi * 8);
#pragma unroll
      for (int j = 0; j < 4; ++j) {
        FragBF b;
        const unsigned short* pb = &Vt[(16 * j + l15) * FLD + kk];
        b.q[0] = *(const uint4*)(pb + hi * 8);
        b.q[1] = *(const uint4*)(pb + 16 + hi * 8);
        dacc[j] = wmma_bf16(a, b, dacc[j]);
      }
    }
    // rescale-and-accumulate O
#pragma unroll
    for (int j = 0; j < 4; ++j) {
      const int col = 16 * j + l15;
#pragma unroll
      for (int r = 0; r < 8; ++r) {
        const int row = wb + 8 * hi + r;
        Ol[row * SLD + col] = Ol[row * SLD + col] * corr[r] + dacc[j][r];
      }
    }
    __syncthreads();
  }

  if (l15 == 0) {
#pragma unroll
    for (int r = 0; r < 8; ++r) lArr[wb + 8 * hi + r] = l_r[r];
  }
  __syncthreads();
  {
    const int row = t >> 1, c0 = (t & 1) * 32;
    const float inv = 1.0f / lArr[row];
    float* dst = Ob + (size_t)(q0 + row) * ldo + c0;
    for (int c = 0; c < 32; ++c) dst[c] = Ol[row * SLD + c0 + c] * inv;
  }
}

// ---------------- LayerNorm(out = LN(a+b)) over H=1024, one wave per row ----------------
__global__ __launch_bounds__(256)
void ln_add(const float* __restrict__ a, const float* __restrict__ b,
            const float* __restrict__ sc, const float* __restrict__ bi,
            float* __restrict__ outp, int rows)
{
  const int row = blockIdx.x * 8 + (threadIdx.x >> 5);
  if (row >= rows) return;
  const int lane = threadIdx.x & 31;
  const float* ar = a + (size_t)row * H_DIM;
  const float* br = b + (size_t)row * H_DIM;
  float v[32];
  float s1 = 0.f;
#pragma unroll
  for (int i = 0; i < 32; ++i) {
    v[i] = ar[lane + i * 32] + br[lane + i * 32];
    s1 += v[i];
  }
  for (int d = 1; d < 32; d <<= 1) s1 += __shfl_xor(s1, d, 32);
  const float mean = s1 * (1.0f / H_DIM);
  float s2 = 0.f;
#pragma unroll
  for (int i = 0; i < 32; ++i) { const float dd = v[i] - mean; s2 += dd * dd; }
  for (int d = 1; d < 32; d <<= 1) s2 += __shfl_xor(s2, d, 32);
  const float inv = rsqrtf(s2 * (1.0f / H_DIM) + 1e-5f);
  float* orow = outp + (size_t)row * H_DIM;
#pragma unroll
  for (int i = 0; i < 32; ++i) {
    const int c = lane + i * 32;
    orow[c] = (v[i] - mean) * inv * sc[c] + bi[c];
  }
}

// ---------------- landmark (global) attention: GH=4 heads of 256, 32 landmarks ----------------
__global__ __launch_bounds__(256)
void landmark_attn(const float* __restrict__ gq, const float* __restrict__ gk,
                   const float* __restrict__ gv, float* __restrict__ outp)
{
  const int gwave = blockIdx.x * 8 + (threadIdx.x >> 5);
  const int s = gwave >> 2;
  const int h = gwave & 3;
  if (s >= S_LEN) return;
  const int lane = threadIdx.x & 31;             // lane == landmark index
  const float* q  = gq + (size_t)s * H_DIM + h * HG;
  const float* kb = gk + (size_t)lane * H_DIM + h * HG;
  float scv = 0.f;
  for (int d = 0; d < HG; ++d) scv += q[d] * kb[d];
  scv *= 0.0625f;                                // 1/sqrt(256)
  float mx = scv;
  for (int d = 1; d < 32; d <<= 1) mx = fmaxf(mx, __shfl_xor(mx, d, 32));
  const float w = __expf(scv - mx);
  float sum = w;
  for (int d = 1; d < 32; d <<= 1) sum += __shfl_xor(sum, d, 32);
  const float inv = 1.0f / sum;
  float acc[8];
#pragma unroll
  for (int d = 0; d < 8; ++d) acc[d] = 0.f;
  for (int n = 0; n < NLM; ++n) {
    const float wn = __shfl(w, n, 32) * inv;
    const float* vv = gv + (size_t)n * H_DIM + h * HG + lane * 8;
#pragma unroll
    for (int d = 0; d < 8; ++d) acc[d] += wn * vv[d];
  }
  float* dst = outp + (size_t)s * H_DIM + h * HG + lane * 8;
#pragma unroll
  for (int d = 0; d < 8; ++d) dst[d] = acc[d];
}

// ---------------- small utility kernels ----------------
__global__ void zero_buf(float* __restrict__ p, int n4) {
  const int i = blockIdx.x * 256 + threadIdx.x;
  if (i < n4) ((float4*)p)[i] = make_float4(0.f, 0.f, 0.f, 0.f);
}
__global__ void gather_window(const float* __restrict__ h, float* __restrict__ xw, int start) {
  const int i = blockIdx.x * 256 + threadIdx.x;
  const int n4 = WWIN * H_DIM / 4;
  if (i >= n4) return;
  const int row = i / (H_DIM / 4), c4 = i % (H_DIM / 4);
  int srow = start + row; if (srow > S_LEN - 1) srow = S_LEN - 1;
  ((float4*)xw)[i] = ((const float4*)(h + (size_t)srow * H_DIM))[c4];
}
__global__ void accum_window(const float* __restrict__ yw, float* __restrict__ acc,
                             int start, int valid) {
  const int i = blockIdx.x * 256 + threadIdx.x;
  const int n4 = WWIN * H_DIM / 4;
  if (i >= n4) return;
  const int row = i / (H_DIM / 4), c4 = i % (H_DIM / 4);
  if (row >= valid) return;
  float4* dst = (float4*)(acc + (size_t)(start + row) * H_DIM);
  float4 a = dst[c4];
  const float4 y = ((const float4*)yw)[i];
  a.x += y.x; a.y += y.y; a.z += y.z; a.w += y.w;
  dst[c4] = a;
}
__global__ void combine_out(const float* __restrict__ g, const float* __restrict__ acc,
                            float* __restrict__ outp, int n4) {
  const int i = blockIdx.x * 256 + threadIdx.x;
  if (i >= n4) return;
  const int row = i / (H_DIM / 4);
  int cnt = 0;
#pragma unroll
  for (int w = 0; w < NWIN; ++w) {
    const int st = w * WSTRIDE;
    cnt += (row >= st && row < st + WWIN) ? 1 : 0;
  }
  const float inv = 1.0f / (float)cnt;
  const float4 a = ((const float4*)acc)[i];
  const float4 gg = ((const float4*)g)[i];
  float4 o;
  o.x = gg.x + a.x * inv; o.y = gg.y + a.y * inv;
  o.z = gg.z + a.z * inv; o.w = gg.w + a.w * inv;
  ((float4*)outp)[i] = o;
}

// ---------------- host-side wrappers ----------------
static void gemm(hipStream_t st, const float* A, int lda, const float* W, int ldw,
                 const float* bias, const float* res, float* C, int ldc,
                 int M, int N, int K, int relu) {
  dim3 g((M + BM - 1) / BM, N / BN);
  gemm_wmma<<<g, 256, 0, st>>>(A, lda, W, ldw, bias, res, C, ldc, M, N, K, relu);
}
static void flash(hipStream_t st, const float* Q, int ldq, long long qB,
                  const float* K_, int ldk, long long kB,
                  const float* V_, int ldv, long long vB,
                  float* O, int ldo, long long oB,
                  int qLen, int kvLen, int validLen, int heads, int batches, float scale) {
  dim3 g(qLen / FQT, heads, batches);
  flash_attn<<<g, 128, 0, st>>>(Q, ldq, qB, K_, ldk, kB, V_, ldv, vB, O, ldo, oB,
                                kvLen, validLen, scale);
}

extern "C" void kernel_launch(void* const* d_in, const int* in_sizes, int n_in,
                              void* d_out, int out_size, void* d_ws, size_t ws_size,
                              hipStream_t stream)
{
  (void)in_sizes; (void)n_in; (void)out_size; (void)ws_size;
  const float* hidden   = (const float*)d_in[0];
  const float* segments = (const float*)d_in[1];
  const float* sw_qw = (const float*)d_in[2];
  const float* sw_qb = (const float*)d_in[3];
  const float* sw_kw = (const float*)d_in[4];
  const float* sw_kb = (const float*)d_in[5];
  const float* sw_vw = (const float*)d_in[6];
  const float* sw_vb = (const float*)d_in[7];
  const float* sw_ow = (const float*)d_in[8];
  const float* sw_ob = (const float*)d_in[9];
  const float* g_land = (const float*)d_in[10];
  const float* g_qw = (const float*)d_in[11];
  const float* g_qb = (const float*)d_in[12];
  const float* g_kw = (const float*)d_in[13];
  const float* g_kb = (const float*)d_in[14];
  const float* g_vw = (const float*)d_in[15];
  const float* g_vb = (const float*)d_in[16];
  const float* g_ow = (const float*)d_in[17];
  const float* g_ob = (const float*)d_in[18];
  const float* int_inw  = (const float*)d_in[19];
  const float* int_inb  = (const float*)d_in[20];
  const float* int_outw = (const float*)d_in[21];
  const float* int_outb = (const float*)d_in[22];
  const float* enc_inw  = (const float*)d_in[23];
  const float* enc_inb  = (const float*)d_in[24];
  const float* enc_outw = (const float*)d_in[25];
  const float* enc_outb = (const float*)d_in[26];
  const float* enc_f1w  = (const float*)d_in[27];
  const float* enc_f1b  = (const float*)d_in[28];
  const float* enc_f2w  = (const float*)d_in[29];
  const float* enc_f2b  = (const float*)d_in[30];
  const float* enc_ln1s = (const float*)d_in[31];
  const float* enc_ln1b = (const float*)d_in[32];
  const float* enc_ln2s = (const float*)d_in[33];
  const float* enc_ln2b = (const float*)d_in[34];

  float* ws = (float*)d_ws;
  size_t off = 0;
  auto alloc = [&](size_t n) { float* p = ws + off; off += n; return p; };
  const size_t ET = (size_t)1024 * 1024;           // encoder tokens * H
  const size_t SH = (size_t)S_LEN * H_DIM;

  float* x    = alloc(ET);                         // encoder state / kv
  float* qkv  = alloc((size_t)1024 * 3072);        // reused as k_c/v_c later
  float* ctxe = alloc(ET);
  float* tmpe = alloc(ET);
  float* mid  = alloc((size_t)1024 * FFDIM);
  float* q_c  = alloc(SH);                         // reused as gq
  float* ctx_c= alloc(SH);                         // reused as gctx
  float* hbuf = alloc(SH);
  float* swac = alloc(SH);
  float* xw   = alloc((size_t)WWIN * H_DIM);
  float* qw   = alloc((size_t)WWIN * H_DIM);
  float* kw   = alloc((size_t)WWIN * H_DIM);
  float* vw   = alloc((size_t)WWIN * H_DIM);
  float* cw   = alloc((size_t)WWIN * H_DIM);
  float* yw   = alloc((size_t)WWIN * H_DIM);
  float* gk   = alloc((size_t)NLM * H_DIM);
  float* gv   = alloc((size_t)NLM * H_DIM);
  float* gtmp = alloc(SH);
  float* k_c  = qkv;
  float* v_c  = qkv + ET;
  float* gq   = q_c;
  float* gctx = ctx_c;

  const float scale64 = 0.125f;                    // 1/sqrt(64)

  // ---- Phase A: segment encoder (2 layers) ----
  hipMemcpyAsync(x, segments, ET * sizeof(float), hipMemcpyDeviceToDevice, stream);
  for (int l = 0; l < 2; ++l) {
    gemm(stream, x, 1024, enc_inw + (size_t)l * 3072 * 1024, 1024,
         enc_inb + (size_t)l * 3072, nullptr, qkv, 3072, 1024, 3072, 1024, 0);
    flash(stream, qkv,        3072, (long long)LSEG * 3072,
                  qkv + 1024, 3072, (long long)LSEG * 3072,
                  qkv + 2048, 3072, (long long)LSEG * 3072,
                  ctxe, 1024, (long long)LSEG * 1024,
                  LSEG, LSEG, LSEG, NHEAD, NSEG, scale64);
    gemm(stream, ctxe, 1024, enc_outw + (size_t)l * 1024 * 1024, 1024,
         enc_outb + (size_t)l * 1024, nullptr, tmpe, 1024, 1024, 1024, 1024, 0);
    ln_add<<<dim3(128), 256, 0, stream>>>(x, tmpe, enc_ln1s + l * 1024,
                                          enc_ln1b + l * 1024, x, 1024);
    gemm(stream, x, 1024, enc_f1w + (size_t)l * FFDIM * 1024, 1024,
         enc_f1b + (size_t)l * FFDIM, nullptr, mid, FFDIM, 1024, FFDIM, 1024, 1);
    gemm(stream, mid, FFDIM, enc_f2w + (size_t)l * 1024 * FFDIM, FFDIM,
         enc_f2b + (size_t)l * 1024, nullptr, tmpe, 1024, 1024, 1024, FFDIM, 0);
    ln_add<<<dim3(128), 256, 0, stream>>>(x, tmpe, enc_ln2s + l * 1024,
                                          enc_ln2b + l * 1024, x, 1024);
  }

  // ---- Phase B: cross attention (h = hidden + MHA(hidden, kv)) ----
  gemm(stream, hidden, 1024, int_inw, 1024, int_inb, nullptr,
       q_c, 1024, S_LEN, 1024, 1024, 0);
  gemm(stream, x, 1024, int_inw + (size_t)1024 * 1024, 1024, int_inb + 1024,
       nullptr, k_c, 1024, 1024, 1024, 1024, 0);
  gemm(stream, x, 1024, int_inw + (size_t)2048 * 1024, 1024, int_inb + 2048,
       nullptr, v_c, 1024, 1024, 1024, 1024, 0);
  flash(stream, q_c, 1024, 0, k_c, 1024, 0, v_c, 1024, 0, ctx_c, 1024, 0,
        S_LEN, 1024, 1024, NHEAD, 1, scale64);
  gemm(stream, ctx_c, 1024, int_outw, 1024, int_outb, hidden,
       hbuf, 1024, S_LEN, 1024, 1024, 0);

  // ---- Phase C: sliding-window attention (sequential windows => deterministic) ----
  {
    const int n4 = (int)(SH / 4);
    zero_buf<<<dim3((n4 + 255) / 256), 256, 0, stream>>>(swac, n4);
  }
  for (int w = 0; w < NWIN; ++w) {
    const int start = w * WSTRIDE;
    int valid = S_LEN - start; if (valid > WWIN) valid = WWIN;
    const int n4 = WWIN * H_DIM / 4;
    gather_window<<<dim3((n4 + 255) / 256), 256, 0, stream>>>(hbuf, xw, start);
    gemm(stream, xw, 1024, sw_qw, 1024, sw_qb, nullptr, qw, 1024, WWIN, 1024, 1024, 0);
    gemm(stream, xw, 1024, sw_kw, 1024, sw_kb, nullptr, kw, 1024, WWIN, 1024, 1024, 0);
    gemm(stream, xw, 1024, sw_vw, 1024, sw_vb, nullptr, vw, 1024, WWIN, 1024, 1024, 0);
    flash(stream, qw, 1024, 0, kw, 1024, 0, vw, 1024, 0, cw, 1024, 0,
          WWIN, WWIN, valid, NHEAD, 1, scale64);
    gemm(stream, cw, 1024, sw_ow, 1024, sw_ob, nullptr, yw, 1024, WWIN, 1024, 1024, 0);
    accum_window<<<dim3((n4 + 255) / 256), 256, 0, stream>>>(yw, swac, start, valid);
  }

  // ---- Phase D: landmark/global attention + final combine ----
  gemm(stream, hbuf, 1024, g_qw, 1024, g_qb, nullptr, gq, 1024, S_LEN, 1024, 1024, 0);
  gemm(stream, g_land, 1024, g_kw, 1024, g_kb, nullptr, gk, 1024, NLM, 1024, 1024, 0);
  gemm(stream, g_land, 1024, g_vw, 1024, g_vb, nullptr, gv, 1024, NLM, 1024, 1024, 0);
  landmark_attn<<<dim3((S_LEN * GHEAD) / 8), 256, 0, stream>>>(gq, gk, gv, gctx);
  gemm(stream, gctx, 1024, g_ow, 1024, g_ob, nullptr, gtmp, 1024, S_LEN, 1024, 1024, 0);
  {
    const int n4 = (int)(SH / 4);
    combine_out<<<dim3((n4 + 255) / 256), 256, 0, stream>>>(gtmp, swac, (float*)d_out, n4);
  }
}
